// EnhancedPCNAttention_85246510891336
// MI455X (gfx1250) — compile-verified, matrix operand order checked by reading, not run on hardware
//
#include <hip/hip_runtime.h>
#include <hip/hip_bf16.h>

typedef _Float16 f16;
typedef __attribute__((ext_vector_type(16))) _Float16 v16h;
typedef __attribute__((ext_vector_type(8)))  _Float16 v8h;
typedef __attribute__((ext_vector_type(8)))  float    v8f;

#define BB   2
#define TDIM 2048
#define CDIM 1024
#define NH   16
#define HD   64
#define MT   (BB * TDIM)   // 4096 rows total

// ---------------------------------------------------------------------------
// Fragment loader: both A-frags (lane = M row) and B-frags (lane = N col) of
// v_wmma_f32_16x16x32_f16 read, per lane, two contiguous 8-half (16B) runs of
// a row-major matrix row:  K offsets [kk+8g, kk+8g+8) and [kk+16+8g, ...).
// ---------------------------------------------------------------------------
__device__ __forceinline__ v16h load_frag(const f16* __restrict__ base,
                                          int ld, int kk, int r, int g) {
  const f16* p = base + (size_t)r * ld + kk + g * 8;
  v8h lo = *(const v8h*)(p);
  v8h hi = *(const v8h*)(p + 16);
  v16h out;
#pragma unroll
  for (int j = 0; j < 8; ++j) { out[j] = lo[j]; out[j + 8] = hi[j]; }
  return out;
}

__device__ __forceinline__ v8f wmma16(v16h a, v16h b, v8f c) {
  return __builtin_amdgcn_wmma_f32_16x16x32_f16(false, a, false, b,
                                                (short)0, c, false, false);
}

// ---------------------------------------------------------------------------
// f32 -> f16 conversion
// ---------------------------------------------------------------------------
__global__ void cvt_f32_f16(const float* __restrict__ src,
                            f16* __restrict__ dst, int n) {
  for (int i = blockIdx.x * blockDim.x + threadIdx.x; i < n;
       i += gridDim.x * blockDim.x)
    dst[i] = (f16)src[i];
}

// ---------------------------------------------------------------------------
// Fused QKV projection: Q/K/V = x @ W^T + b.  One wave per 64x16 output tile
// (4 M-tiles x {Q,K,V}): per k-step 7 fragment loads feed 12 WMMAs.
// __launch_bounds__(...,1): ~150 VGPRs/lane needed, run at 1 wave/EU min so
// the accumulators stay in registers (no scratch spills).
// ---------------------------------------------------------------------------
__global__ void __launch_bounds__(256, 1)
qkv_gemm(const f16* __restrict__ xh,
         const f16* __restrict__ wqh,
         const f16* __restrict__ wkh,
         const f16* __restrict__ wvh,
         const float* __restrict__ bq,
         const float* __restrict__ bk,
         const float* __restrict__ bv,
         f16* __restrict__ qh,
         f16* __restrict__ kh,
         f16* __restrict__ vth) {
  const int wave = (blockIdx.x * blockDim.x + threadIdx.x) >> 5;
  const int lane = threadIdx.x & 31;
  const int r = lane & 15, g = lane >> 4;
  const int tm = wave % (MT / 64);      // 64 M-super-tiles of 64 rows
  const int tn = wave / (MT / 64);      // 64 N-tiles of 16 cols

  const f16* arow = xh  + (size_t)(tm * 64) * CDIM;
  const f16* qrow = wqh + (size_t)(tn * 16) * CDIM;
  const f16* krow = wkh + (size_t)(tn * 16) * CDIM;
  const f16* vrow = wvh + (size_t)(tn * 16) * CDIM;

  v8f accq[4] = {}, acck[4] = {}, accv[4] = {};
#pragma unroll 2
  for (int kk = 0; kk < CDIM; kk += 32) {
    v16h bqf = load_frag(qrow, CDIM, kk, r, g);
    v16h bkf = load_frag(krow, CDIM, kk, r, g);
    v16h bvf = load_frag(vrow, CDIM, kk, r, g);
#pragma unroll
    for (int mi = 0; mi < 4; ++mi) {
      v16h a = load_frag(arow + (size_t)(mi * 16) * CDIM, CDIM, kk, r, g);
      accq[mi] = wmma16(a, bqf, accq[mi]);
      acck[mi] = wmma16(a, bkf, acck[mi]);
      accv[mi] = wmma16(a, bvf, accv[mi]);
    }
  }

  const int gc = tn * 16 + r;          // output feature column
  const int h = gc >> 6, d = gc & 63;  // head, dim-in-head
  const float biasq = bq[gc], biask = bk[gc], biasv = bv[gc];
#pragma unroll
  for (int mi = 0; mi < 4; ++mi) {
#pragma unroll
    for (int v = 0; v < 8; ++v) {
      const int m = tm * 64 + mi * 16 + g * 8 + v;   // global row (b,t)
      const int b = m / TDIM, t = m % TDIM;
      const int bh = b * NH + h;
      qh[((size_t)bh * TDIM + t) * HD + d] = (f16)((accq[mi][v] + biasq) * 0.125f);
      kh[((size_t)bh * TDIM + t) * HD + d] = (f16)(acck[mi][v] + biask);
      vth[((size_t)bh * HD + d) * TDIM + t] = (f16)(accv[mi][v] + biasv);
    }
  }
}

// ---------------------------------------------------------------------------
// Causal flash attention.  One wave per (head, PAIR of 16-query tiles) so
// every K/V fragment load feeds two WMMAs.  Scores computed TRANSPOSED
// (S^T = K·Q^T): the softmax'd tile is already in A-frag layout for P·V
// (lane = query row, VGPR elements = key index) -- no LDS round trip.
// ---------------------------------------------------------------------------
__global__ void __launch_bounds__(256, 1)
attn_kernel(const f16* __restrict__ qh,
            const f16* __restrict__ kh,
            const f16* __restrict__ vth,
            f16* __restrict__ ah) {
  const int wave = (blockIdx.x * blockDim.x + threadIdx.x) >> 5;
  const int lane = threadIdx.x & 31;
  const int r = lane & 15, g = lane >> 4;
  const int j  = wave & 63;    // query-tile pair: queries [32j, 32j+32)
  const int bh = wave >> 6;    // head index (0..31)
  const int b = bh >> 4, h = bh & 15;

  const f16* Q  = qh  + (size_t)bh * TDIM * HD;
  const f16* K  = kh  + (size_t)bh * TDIM * HD;
  const f16* VT = vth + (size_t)bh * HD * TDIM;

  // Loop-invariant Q B-fragments for both query tiles (kk = 0 and 32)
  const f16* q0 = Q + (size_t)(j * 32) * HD;
  const f16* q1 = q0 + (size_t)16 * HD;
  const v16h qb00 = load_frag(q0, HD, 0,  r, g);
  const v16h qb01 = load_frag(q0, HD, 32, r, g);
  const v16h qb10 = load_frag(q1, HD, 0,  r, g);
  const v16h qb11 = load_frag(q1, HD, 32, r, g);

  v8f oA[4] = {}, oB[4] = {};          // O accumulators, tile 0 / tile 1
  float mA = -3.0e38f, lA = 0.0f;
  float mB = -3.0e38f, lB = 0.0f;
  const int qgA = j * 32 + r;          // lane's query row, tile 0
  const int qgB = j * 32 + 16 + r;     // lane's query row, tile 1
  const int np = j + 1;                // number of 32-key pairs (causal)

  for (int p = 0; p < np; ++p) {
    const int kb0 = p * 32, kb1 = p * 32 + 16;

    // S^T tiles: A = K rows, B = Q^T.  Each K frag feeds both query tiles.
    v8f s0a = {}, s1a = {}, s0b = {}, s1b = {};
    {
      v16h ka = load_frag(K + (size_t)kb0 * HD, HD, 0, r, g);
      s0a = wmma16(ka, qb00, s0a);  s0b = wmma16(ka, qb10, s0b);
    }
    {
      v16h ka = load_frag(K + (size_t)kb0 * HD, HD, 32, r, g);
      s0a = wmma16(ka, qb01, s0a);  s0b = wmma16(ka, qb11, s0b);
    }
    {
      v16h ka = load_frag(K + (size_t)kb1 * HD, HD, 0, r, g);
      s1a = wmma16(ka, qb00, s1a);  s1b = wmma16(ka, qb10, s1b);
    }
    {
      v16h ka = load_frag(K + (size_t)kb1 * HD, HD, 32, r, g);
      s1a = wmma16(ka, qb01, s1a);  s1b = wmma16(ka, qb11, s1b);
    }

    if (p == np - 1) {                 // causal mask only on the final pair
#pragma unroll
      for (int v = 0; v < 8; ++v) {
        const int k0 = kb0 + g * 8 + v, k1 = kb1 + g * 8 + v;
        if (k0 > qgA) s0a[v] = -3.0e30f;
        if (k1 > qgA) s1a[v] = -3.0e30f;
        if (k0 > qgB) s0b[v] = -3.0e30f;
        if (k1 > qgB) s1b[v] = -3.0e30f;
      }
    }

    // ---- online softmax (lane <-> query), both tiles ----
    float mxA = s0a[0], mxB = s0b[0];
#pragma unroll
    for (int v = 0; v < 8; ++v) {
      mxA = fmaxf(mxA, s0a[v]); mxA = fmaxf(mxA, s1a[v]);
      mxB = fmaxf(mxB, s0b[v]); mxB = fmaxf(mxB, s1b[v]);
    }
    mxA = fmaxf(mxA, __shfl_xor(mxA, 16, 32));
    mxB = fmaxf(mxB, __shfl_xor(mxB, 16, 32));
    const float mnA = fmaxf(mA, mxA), mnB = fmaxf(mB, mxB);
    const float fA = __expf(mA - mnA), fB = __expf(mB - mnB);

    float sumA = 0.0f, sumB = 0.0f;
    v16h paA, paB;                     // P tiles in A-frag layout
#pragma unroll
    for (int v = 0; v < 8; ++v) {
      const float e0a = __expf(s0a[v] - mnA), e1a = __expf(s1a[v] - mnA);
      const float e0b = __expf(s0b[v] - mnB), e1b = __expf(s1b[v] - mnB);
      sumA += e0a + e1a;  sumB += e0b + e1b;
      paA[v] = (f16)e0a;  paA[v + 8] = (f16)e1a;
      paB[v] = (f16)e0b;  paB[v + 8] = (f16)e1b;
    }
    sumA += __shfl_xor(sumA, 16, 32);
    sumB += __shfl_xor(sumB, 16, 32);
    lA = lA * fA + sumA;  lB = lB * fB + sumB;
    mA = mnA;             mB = mnB;

    // rescale O accumulators (broadcast per-query factor to C/D row layout)
#pragma unroll
    for (int v = 0; v < 8; ++v) {
      const float fvA = __shfl(fA, g * 8 + v, 32);
      const float fvB = __shfl(fB, g * 8 + v, 32);
#pragma unroll
      for (int c = 0; c < 4; ++c) { oA[c][v] *= fvA; oB[c][v] *= fvB; }
    }

    // P·V: each V^T B-frag feeds both query tiles
#pragma unroll
    for (int c = 0; c < 4; ++c) {
      v16h vb = load_frag(VT + (size_t)(c * 16) * TDIM, TDIM, p * 32, r, g);
      oA[c] = wmma16(paA, vb, oA[c]);
      oB[c] = wmma16(paB, vb, oB[c]);
    }
  }

  // finalize: divide by row sums and store f16 to [B,T,C]
  const float liA = 1.0f / lA, liB = 1.0f / lB;
#pragma unroll
  for (int v = 0; v < 8; ++v) {
    const float fvA = __shfl(liA, g * 8 + v, 32);
    const float fvB = __shfl(liB, g * 8 + v, 32);
    const int tA = j * 32 + g * 8 + v;
    const int tB = tA + 16;
    f16* dA = ah + ((size_t)b * TDIM + tA) * CDIM + h * HD;
    f16* dB = ah + ((size_t)b * TDIM + tB) * CDIM + h * HD;
#pragma unroll
    for (int c = 0; c < 4; ++c) {
      dA[c * 16 + r] = (f16)(oA[c][v] * fvA);
      dB[c * 16 + r] = (f16)(oB[c][v] * fvB);
    }
  }
}

// ---------------------------------------------------------------------------
// Output projection: out = ah @ wo^T + bo, f32 result.  2Mx4N register
// blocking: 6 fragment loads feed 8 WMMAs per k-step.
// ---------------------------------------------------------------------------
__global__ void __launch_bounds__(256, 1)
out_gemm(const f16* __restrict__ ahx,
         const f16* __restrict__ woh,
         const float* __restrict__ bo,
         float* __restrict__ out) {
  const int wave = (blockIdx.x * blockDim.x + threadIdx.x) >> 5;
  const int lane = threadIdx.x & 31;
  const int r = lane & 15, g = lane >> 4;
  const int tm = wave % (MT / 32);      // 128 M-super-tiles of 32 rows
  const int tn = wave / (MT / 32);      // 16 N-super-tiles of 64 cols

  const f16* arow = ahx + (size_t)(tm * 32) * CDIM;
  const f16* brow = woh + (size_t)(tn * 64) * CDIM;

  v8f acc[2][4] = {};
#pragma unroll 2
  for (int kk = 0; kk < CDIM; kk += 32) {
    v16h a0 = load_frag(arow, CDIM, kk, r, g);
    v16h a1 = load_frag(arow + (size_t)16 * CDIM, CDIM, kk, r, g);
#pragma unroll
    for (int ni = 0; ni < 4; ++ni) {
      v16h bf = load_frag(brow + (size_t)(ni * 16) * CDIM, CDIM, kk, r, g);
      acc[0][ni] = wmma16(a0, bf, acc[0][ni]);
      acc[1][ni] = wmma16(a1, bf, acc[1][ni]);
    }
  }

#pragma unroll
  for (int ni = 0; ni < 4; ++ni) {
    const int gc = tn * 64 + ni * 16 + r;
    const float bias = bo[gc];
#pragma unroll
    for (int mi = 0; mi < 2; ++mi)
#pragma unroll
      for (int v = 0; v < 8; ++v)
        out[(size_t)(tm * 32 + mi * 16 + g * 8 + v) * CDIM + gc] =
            acc[mi][ni][v] + bias;
  }
}

// ---------------------------------------------------------------------------
extern "C" void kernel_launch(void* const* d_in, const int* in_sizes, int n_in,
                              void* d_out, int out_size, void* d_ws, size_t ws_size,
                              hipStream_t stream) {
  const float* x  = (const float*)d_in[0];
  const float* wq = (const float*)d_in[1];
  const float* bq = (const float*)d_in[2];
  const float* wk = (const float*)d_in[3];
  const float* bk = (const float*)d_in[4];
  const float* wv = (const float*)d_in[5];
  const float* bv = (const float*)d_in[6];
  const float* wo = (const float*)d_in[7];
  const float* bo = (const float*)d_in[8];
  // d_in[9] = causal mask (implemented analytically)

  f16* ws  = (f16*)d_ws;
  f16* xh  = ws;                               // MT*C
  f16* wqh = xh  + (size_t)MT * CDIM;          // C*C
  f16* wkh = wqh + (size_t)CDIM * CDIM;
  f16* wvh = wkh + (size_t)CDIM * CDIM;
  f16* woh = wvh + (size_t)CDIM * CDIM;
  f16* qh  = woh + (size_t)CDIM * CDIM;        // MT*C  (per-head [T][D])
  f16* kh  = qh  + (size_t)MT * CDIM;
  f16* vth = kh  + (size_t)MT * CDIM;          // per-head transposed [D][T]
  f16* ah  = vth + (size_t)MT * CDIM;          // attention output [B*T][C]

  const int cb = 256;
  cvt_f32_f16<<<(MT * CDIM + cb - 1) / cb, cb, 0, stream>>>(x,  xh,  MT * CDIM);
  cvt_f32_f16<<<(CDIM * CDIM + cb - 1) / cb, cb, 0, stream>>>(wq, wqh, CDIM * CDIM);
  cvt_f32_f16<<<(CDIM * CDIM + cb - 1) / cb, cb, 0, stream>>>(wk, wkh, CDIM * CDIM);
  cvt_f32_f16<<<(CDIM * CDIM + cb - 1) / cb, cb, 0, stream>>>(wv, wvh, CDIM * CDIM);
  cvt_f32_f16<<<(CDIM * CDIM + cb - 1) / cb, cb, 0, stream>>>(wo, woh, CDIM * CDIM);

  // (MT/64)*(CDIM/16) = 4096 waves, 8 waves (256 threads) per block
  qkv_gemm<<<4096 / 8, 256, 0, stream>>>(xh, wqh, wkh, wvh, bq, bk, bv,
                                         qh, kh, vth);

  // B*H*(T/32) = 2048 waves
  attn_kernel<<<2048 / 8, 256, 0, stream>>>(qh, kh, vth, ah);

  // (MT/32)*(CDIM/64) = 2048 waves
  out_gemm<<<2048 / 8, 256, 0, stream>>>(ah, woh, bo, (float*)d_out);
}
